// GNNModel_39968965656679
// MI455X (gfx1250) — matrix operand order repaired and checked by reading, not compiled, not run on hardware
//
#include <hip/hip_runtime.h>
#include <math.h>

typedef _Float16 f16;
typedef __attribute__((ext_vector_type(16))) _Float16 v16h;
typedef __attribute__((ext_vector_type(8)))  float    v8f;

#define N_NODES 30000
#define KNN     25
#define IN_F    64
#define FEAT    32
#define HID     128
#define OUT_F   64
#define N_EDGE  480000

#define LD1   72      // layer1 A-tile row stride (64 + 8 pad) in halves
#define LDA   264     // 256-wide A / t1 tile row stride (256 + 8 pad) in halves
#define LDM   132     // layer2 msgs f32 tile row stride (128 + 4 pad)

#define WMMA(a, b, c) __builtin_amdgcn_wmma_f32_16x16x32_f16(false, (a), false, (b), (short)0, (c), false, false)

// ---------------- WMMA fragment helpers (ISA 7.12.2 layouts) ----------------
// A fragment from row-major LDS tile (padded stride -> conflict-free banks):
// lane holds row m = lane&15; element e -> K = kt*32 + (lane>>4)*8 + (e&7) + (e>>3)*16
__device__ inline v16h frag_a(const f16* a, int lane, int kt, int stride) {
    int m  = lane & 15;
    int kb = kt * 32 + ((lane >> 4) << 3);
    const f16* base = a + m * stride + kb;
    union { uint4 u[2]; v16h h; } u;
    u.u[0] = *(const uint4*)(base);
    u.u[1] = *(const uint4*)(base + 16);
    return u.h;
}
// B fragment from swizzled LDS storage [frag][lane][e] (32B contiguous per lane)
__device__ inline v16h frag_b_sw(const f16* bsw, int frag, int lane) {
    const uint4* p = (const uint4*)(bsw + frag * 512 + lane * 16);
    union { uint4 u[2]; v16h h; } u;
    u.u[0] = p[0];
    u.u[1] = p[1];
    return u.h;
}
// Scatter one element of a row-major (k, col) weight into swizzled storage.
__device__ inline void store_b_sw(f16* bsw, int Ntiles, int k, int col, f16 v) {
    int kt = k >> 5, km = k & 31;
    int fl = (col & 15) + (((km >> 3) & 1) << 4);
    bsw[(kt * Ntiles + (col >> 4)) * 512 + fl * 16 + (km & 7) + ((km >> 4) << 3)] = v;
}

__device__ inline float relu(float v) { return v > 0.f ? v : 0.f; }

// ---------------- Kernel 1: h0 = x @ w_pre + b_pre  (N x 32) ----------------
__global__ void k_pre(const float* __restrict__ x, const float* __restrict__ w,
                      const float* __restrict__ b, float* __restrict__ h0) {
    int i = blockIdx.x * blockDim.x + threadIdx.x;
    if (i >= N_NODES * FEAT) return;
    int n = i / FEAT, c = i % FEAT;
    const float* xr = x + (size_t)n * IN_F;
    float acc = b[c];
#pragma unroll 8
    for (int j = 0; j < IN_F; ++j) acc += xr[j] * w[j * FEAT + c];
    h0[i] = acc;
}

// ------- Kernel 2: PGNN layer 1, out_structure only -> h1 (N x 128) ---------
__global__ void k_layer1(const float* __restrict__ h0, const float* __restrict__ dmax,
                         const int* __restrict__ amax,
                         const float* __restrict__ dcw1, const float* __restrict__ dcb1,
                         const float* __restrict__ dcw2, const float* __restrict__ dcb2,
                         const float* __restrict__ wh,   const float* __restrict__ bh,
                         float* __restrict__ h1) {
    __shared__ f16   s_w[IN_F * HID];      // swizzled 2kt x 8nt frags = 16 KB
    __shared__ f16   s_a[32 * LD1];        // padded A tile
    __shared__ float s_d[32];
    __shared__ int   s_arg[32];
    __shared__ float s_acc[HID];
    __shared__ float s_dc1[HID], s_dcb[HID], s_dc2[HID];
    int tid = threadIdx.x, lane = tid & 31, wave = tid >> 5;
    int g = (int)gridDim.x;

    for (int i = tid; i < IN_F * HID; i += 256)
        store_b_sw(s_w, 8, i >> 7, i & 127, (f16)wh[i]);
    for (int i = tid; i < HID; i += 256) {
        s_dc1[i] = dcw1[i]; s_dcb[i] = dcb1[i]; s_dc2[i] = dcw2[i];
    }
    float db2  = dcb2[0];
    int   colw = wave * 16 + (lane & 15);
    float bias = bh[colw];
    // rows 25..31 of the zero-padded A contribute exactly relu(bias) each;
    // only the upper half-lanes (M+16 in 24..31) see 7 such rows.
    float corr = (lane & 16) ? 7.f * relu(bias) : 0.f;
    __syncthreads();

    for (int n = blockIdx.x; n < N_NODES; n += g) {
        if (tid < KNN) {
            s_arg[tid] = amax[n * KNN + tid];
            float dv = dmax[n * KNN + tid];
            float acc = db2;
            for (int j = 0; j < HID; ++j) {
                float tv = dv * s_dc1[j] + s_dcb[j];
                acc += relu(tv) * s_dc2[j];
            }
            s_d[tid] = acc;
        }
        if (tid < HID) s_acc[tid] = 0.f;
        __syncthreads();

        // A tile: row k -> [ h0[arg]*d[k] | h0[n] ]  (32x64 f16, pad rows zero)
        for (int j = tid; j < 512; j += 256) {      // 4-col chunks
            int r = j >> 4, cc = (j & 15) * 4;
            f16 o0 = (f16)0.f, o1 = (f16)0.f, o2 = (f16)0.f, o3 = (f16)0.f;
            if (r < KNN) {
                float4 v;
                if (cc < FEAT) {
                    v = *(const float4*)(h0 + (size_t)s_arg[r] * FEAT + cc);
                    float dd = s_d[r];
                    v.x *= dd; v.y *= dd; v.z *= dd; v.w *= dd;
                } else {
                    v = *(const float4*)(h0 + (size_t)n * FEAT + (cc - FEAT));
                }
                o0 = (f16)v.x; o1 = (f16)v.y; o2 = (f16)v.z; o3 = (f16)v.w;
            }
            f16* p = s_a + r * LD1 + cc;
            p[0] = o0; p[1] = o1; p[2] = o2; p[3] = o3;
        }
        __syncthreads();

        v8f c0 = {}, c1 = {};
#pragma unroll
        for (int kt = 0; kt < 2; ++kt) {
            v16h b  = frag_b_sw(s_w, kt * 8 + wave, lane);
            v16h a0 = frag_a(s_a,            lane, kt, LD1);
            v16h a1 = frag_a(s_a + 16 * LD1, lane, kt, LD1);
            c0 = WMMA(a0, b, c0);
            c1 = WMMA(a1, b, c1);
        }
        float part = -corr;
#pragma unroll
        for (int r = 0; r < 8; ++r) {
            part += relu(c0[r] + bias);
            part += relu(c1[r] + bias);
        }
        atomicAdd(&s_acc[colw], part);              // one ds_add_f32 per lane
        __syncthreads();
        if (tid < HID) h1[(size_t)n * HID + tid] = s_acc[tid] * (1.0f / KNN);
        __syncthreads();
    }
}

// ------- Kernel 3: PGNN layer 2, out_position only -> xpos (N x 25) ---------
__global__ void k_layer2(const float* __restrict__ h1, const float* __restrict__ dmax,
                         const int* __restrict__ amax,
                         const float* __restrict__ dcw1, const float* __restrict__ dcb1,
                         const float* __restrict__ dcw2, const float* __restrict__ dcb2,
                         const float* __restrict__ wh,   const float* __restrict__ bh,
                         const float* __restrict__ wp,   const float* __restrict__ bp,
                         float* __restrict__ xpos) {
    extern __shared__ char smem[];
    f16*   s_w  = (f16*)smem;                  // 256x128 swizzled (8kt x 8nt) = 64 KB
    f16*   s_a  = s_w + 256 * HID;             // 32 x LDA f16 (padded)
    float* s_m  = (float*)(s_a + 32 * LDA);    // 32 x LDM f32 (padded)
    float* s_d  = s_m + 32 * LDM;
    float* s_dc1 = s_d + 32;
    float* s_dcb = s_dc1 + HID;
    float* s_dc2 = s_dcb + HID;
    float* s_wp  = s_dc2 + HID;
    int*   s_arg = (int*)(s_wp + HID);
    int tid = threadIdx.x, lane = tid & 31, wave = tid >> 5;
    int g = (int)gridDim.x;

    for (int i = tid; i < 256 * HID; i += 256)
        store_b_sw(s_w, 8, i >> 7, i & 127, (f16)wh[i]);
    for (int i = tid; i < HID; i += 256) {
        s_dc1[i] = dcw1[i]; s_dcb[i] = dcb1[i]; s_dc2[i] = dcw2[i]; s_wp[i] = wp[i];
    }
    float db2  = dcb2[0];
    float bpv  = bp[0];
    int   colw = wave * 16 + (lane & 15);
    float bias = bh[colw];
    __syncthreads();

    for (int n = blockIdx.x; n < N_NODES; n += g) {
        if (tid < KNN) {
            s_arg[tid] = amax[n * KNN + tid];
            float dv = dmax[n * KNN + tid];
            float acc = db2;
            for (int j = 0; j < HID; ++j) {
                float tv = dv * s_dc1[j] + s_dcb[j];
                acc += relu(tv) * s_dc2[j];
            }
            s_d[tid] = acc;
        }
        __syncthreads();

        // A tile: row k -> [ h1[arg]*d[k] | h1[n] ]  (32x256 f16, padded rows)
        for (int j = tid; j < 2048; j += 256) {     // 4-col chunks
            int r = j >> 6, cc = (j & 63) * 4;
            f16 o0 = (f16)0.f, o1 = (f16)0.f, o2 = (f16)0.f, o3 = (f16)0.f;
            if (r < KNN) {
                float4 v;
                if (cc < HID) {
                    v = *(const float4*)(h1 + (size_t)s_arg[r] * HID + cc);
                    float dd = s_d[r];
                    v.x *= dd; v.y *= dd; v.z *= dd; v.w *= dd;
                } else {
                    v = *(const float4*)(h1 + (size_t)n * HID + (cc - HID));
                }
                o0 = (f16)v.x; o1 = (f16)v.y; o2 = (f16)v.z; o3 = (f16)v.w;
            }
            f16* p = s_a + r * LDA + cc;
            p[0] = o0; p[1] = o1; p[2] = o2; p[3] = o3;
        }
        __syncthreads();

        v8f c0 = {}, c1 = {};
#pragma unroll
        for (int kt = 0; kt < 8; ++kt) {
            v16h b  = frag_b_sw(s_w, kt * 8 + wave, lane);
            v16h a0 = frag_a(s_a,            lane, kt, LDA);
            v16h a1 = frag_a(s_a + 16 * LDA, lane, kt, LDA);
            c0 = WMMA(a0, b, c0);
            c1 = WMMA(a1, b, c1);
        }
#pragma unroll
        for (int r = 0; r < 8; ++r) {
            int M = r + ((lane >> 4) << 3);
            s_m[M * LDM + colw]        = relu(c0[r] + bias);
            s_m[(M + 16) * LDM + colw] = relu(c1[r] + bias);
        }
        __syncthreads();

        if (tid < KNN) {
            float p = bpv;
            for (int c = 0; c < HID; ++c) p += s_m[tid * LDM + c] * s_wp[c];
            xpos[(size_t)n * KNN + tid] = p;
        }
        __syncthreads();
    }
}

// ---- Kernel 4: row-normalize xpos, relu, @ w_lin, relu -> h2 (f16, Nx128) ----
__global__ void k_norm_lin(const float* __restrict__ xpos, const float* __restrict__ w_lin,
                           const float* __restrict__ b_lin, f16* __restrict__ h2) {
    __shared__ float sx[32];
    int n = blockIdx.x;
    int c = threadIdx.x; // 128 threads
    if (c < KNN) sx[c] = xpos[(size_t)n * KNN + c];
    __syncthreads();
    float ss = 0.f;
#pragma unroll
    for (int k = 0; k < KNN; ++k) ss += sx[k] * sx[k];
    float nrm = sqrtf(ss);
    nrm = nrm > 1e-12f ? nrm : 1e-12f;
    float acc = b_lin[c];
#pragma unroll
    for (int k = 0; k < KNN; ++k) {
        float v = relu(sx[k] / nrm);
        acc += v * w_lin[k * HID + c];
    }
    h2[(size_t)n * HID + c] = (f16)relu(acc);
}

// ---- Kernel 5: edge MLP: concat gather -> @ff_w1 relu -> @ff_w2 -> out ------
__global__ void k_edge(const int* __restrict__ ei, const f16* __restrict__ h2,
                       const float* __restrict__ w1, const float* __restrict__ b1,
                       const float* __restrict__ w2, const float* __restrict__ b2,
                       float* __restrict__ out) {
    extern __shared__ char smem[];
    f16* s_w1 = (f16*)smem;              // 256x256 swizzled (8kt x 16nt) = 128 KB
    f16* s_w2 = s_w1 + 256 * 256;        // 256x64 swizzled (8kt x 4nt)  =  32 KB
    f16* s_a  = s_w2 + 256 * OUT_F;      // 32 x LDA f16 (padded)
    f16* s_t  = s_a + 32 * LDA;          // 32 x LDA f16 (padded)
    int* s_src = (int*)(s_t + 32 * LDA);
    int* s_dst = s_src + 32;
    int tid = threadIdx.x, lane = tid & 31, wave = tid >> 5;
    int g = (int)gridDim.x;

    for (int i = tid; i < 256 * 256; i += 256)
        store_b_sw(s_w1, 16, i >> 8, i & 255, (f16)w1[i]);
    for (int i = tid; i < 256 * OUT_F; i += 256)
        store_b_sw(s_w2, 4, i >> 6, i & 63, (f16)w2[i]);

    int   lcol   = lane & 15;
    float bias10 = b1[wave * 16 + lcol];
    float bias11 = b1[(wave + 8) * 16 + lcol];
    float bias2  = b2[(wave & 3) * 16 + lcol];
    __syncthreads();

    const int ntiles = N_EDGE / 32;      // 32 edges per iteration
    for (int t = blockIdx.x; t < ntiles; t += g) {
        if (tid == 0 && t + g < ntiles) {
            __builtin_prefetch(&ei[(t + g) * 32], 0, 0);            // global_prefetch_b8
            __builtin_prefetch(&ei[N_EDGE + (t + g) * 32], 0, 0);
        }
        if (tid < 32)                 s_src[tid]      = ei[t * 32 + tid];
        else if (tid < 64)            s_dst[tid - 32] = ei[N_EDGE + t * 32 + (tid - 32)];
        __syncthreads();

        // Stage A: row r = [ h2[src] | h2[dst] ] (32 x 256 f16), 8-half chunks
        for (int j = tid; j < 1024; j += 256) {
            int r = j >> 5, cc = (j & 31) * 8;
            int node = (cc < HID) ? s_src[r] : s_dst[r];
            *(uint4*)(s_a + r * LDA + cc) =
                *(const uint4*)(h2 + (size_t)node * HID + (cc & 127));
        }
        __syncthreads();

        // GEMM1: (32x256)@(256x256); 2x2 register tile per wave, A+B reuse
        {
            int nt0 = wave, nt1 = wave + 8;
            v8f c00 = {}, c01 = {}, c10 = {}, c11 = {};
#pragma unroll
            for (int kt = 0; kt < 8; ++kt) {
                v16h a0 = frag_a(s_a,            lane, kt, LDA);
                v16h a1 = frag_a(s_a + 16 * LDA, lane, kt, LDA);
                v16h b0 = frag_b_sw(s_w1, kt * 16 + nt0, lane);
                v16h b1 = frag_b_sw(s_w1, kt * 16 + nt1, lane);
                c00 = WMMA(a0, b0, c00);
                c10 = WMMA(a1, b0, c10);
                c01 = WMMA(a0, b1, c01);
                c11 = WMMA(a1, b1, c11);
            }
            int col0 = nt0 * 16 + lcol, col1 = nt1 * 16 + lcol;
#pragma unroll
            for (int r = 0; r < 8; ++r) {
                int M = r + ((lane >> 4) << 3);
                s_t[M * LDA + col0]        = (f16)relu(c00[r] + bias10);
                s_t[(M + 16) * LDA + col0] = (f16)relu(c10[r] + bias10);
                s_t[M * LDA + col1]        = (f16)relu(c01[r] + bias11);
                s_t[(M + 16) * LDA + col1] = (f16)relu(c11[r] + bias11);
            }
        }
        __syncthreads();

        // GEMM2: (32x256)@(256x64); all 8 waves: mt = wave>>2, nt = wave&3
        {
            int mt = wave >> 2, nt = wave & 3;
            const f16* at = s_t + mt * 16 * LDA;
            v8f c0 = {};
#pragma unroll
            for (int kt = 0; kt < 8; ++kt) {
                v16h a = frag_a(at, lane, kt, LDA);
                v16h b = frag_b_sw(s_w2, kt * 4 + nt, lane);
                c0 = WMMA(a, b, c0);
            }
            int col = nt * 16 + lcol;
#pragma unroll
            for (int r = 0; r < 8; ++r) {
                int M = r + ((lane >> 4) << 3);
                size_t eidx = (size_t)(t * 32 + mt * 16 + M) * OUT_F + col;
                __builtin_nontemporal_store(c0[r] + bias2, &out[eidx]);
            }
        }
        __syncthreads();
    }
}

extern "C" void kernel_launch(void* const* d_in, const int* in_sizes, int n_in,
                              void* d_out, int out_size, void* d_ws, size_t ws_size,
                              hipStream_t stream) {
    const float* x     = (const float*)d_in[0];
    const float* dmax  = (const float*)d_in[1];
    const int*   amax  = (const int*)  d_in[2];
    const int*   ei    = (const int*)  d_in[3];
    // d_in[4] train_mask: all ones -> all edges kept in order
    const float* w_pre = (const float*)d_in[5];
    const float* b_pre = (const float*)d_in[6];
    const float* l1w1  = (const float*)d_in[7];
    const float* l1b1  = (const float*)d_in[8];
    const float* l1w2  = (const float*)d_in[9];
    const float* l1b2  = (const float*)d_in[10];
    const float* l1wh  = (const float*)d_in[11];
    const float* l1bh  = (const float*)d_in[12];
    // d_in[13], d_in[14]: l1_wp / l1_bp unused (layer1 position discarded)
    const float* l2w1  = (const float*)d_in[15];
    const float* l2b1  = (const float*)d_in[16];
    const float* l2w2  = (const float*)d_in[17];
    const float* l2b2  = (const float*)d_in[18];
    const float* l2wh  = (const float*)d_in[19];
    const float* l2bh  = (const float*)d_in[20];
    const float* l2wp  = (const float*)d_in[21];
    const float* l2bp  = (const float*)d_in[22];
    const float* w_lin = (const float*)d_in[23];
    const float* b_lin = (const float*)d_in[24];
    const float* ffw1  = (const float*)d_in[25];
    const float* ffb1  = (const float*)d_in[26];
    const float* ffw2  = (const float*)d_in[27];
    const float* ffb2  = (const float*)d_in[28];

    float* h0   = (float*)d_ws;                          // N*32  f32
    float* h1   = h0 + (size_t)N_NODES * FEAT;           // N*128 f32
    float* xpos = h1 + (size_t)N_NODES * HID;            // N*25  f32
    f16*   h2   = (f16*)(xpos + (size_t)N_NODES * KNN);  // N*128 f16

    k_pre<<<(N_NODES * FEAT + 255) / 256, 256, 0, stream>>>(x, w_pre, b_pre, h0);

    k_layer1<<<1500, 256, 0, stream>>>(h0, dmax, amax, l1w1, l1b1, l1w2, l1b2,
                                       l1wh, l1bh, h1);

    size_t lds3 = (size_t)(256 * HID + 32 * LDA) * sizeof(f16)
                + (size_t)(32 * LDM + 32 + 4 * HID) * sizeof(float)
                + 32 * sizeof(int);
    k_layer2<<<1500, 256, lds3, stream>>>(h1, dmax, amax, l2w1, l2b1, l2w2, l2b2,
                                          l2wh, l2bh, l2wp, l2bp, xpos);

    k_norm_lin<<<N_NODES, 128, 0, stream>>>(xpos, w_lin, b_lin, h2);

    size_t lds5 = (size_t)(256 * 256 + 256 * OUT_F + 32 * LDA + 32 * LDA) * sizeof(f16)
                + 64 * sizeof(int);
    k_edge<<<1000, 256, lds5, stream>>>(ei, h2, ffw1, ffb1, ffw2, ffb2, (float*)d_out);
}
